// SimpleNN_18004502905050
// MI455X (gfx1250) — matrix operand order repaired and checked by reading, compiled, not run on hardware
//
#include <hip/hip_runtime.h>

typedef __attribute__((ext_vector_type(16))) _Float16 v16h;
typedef __attribute__((ext_vector_type(8)))  _Float16 v8h;
typedef __attribute__((ext_vector_type(8)))  float    v8f;
typedef __attribute__((ext_vector_type(4)))  float    v4f;

#define DEVINL __device__ __forceinline__

constexpr int NUM_ADD = 30;
constexpr int IN_DIM  = 784;
constexpr int SIZES[NUM_ADD + 1] = {
  32,31,30,29,28,27,26,25,24,23,22,21,20,19,18,17,
  16,15,14,13,12,11,10,10,10,10,10,10,10,10,10};
constexpr int pad16i(int x) { return (x + 15) & ~15; }

// Strides chosen so (stride in dwords) mod 64 gives >=4-bank row distance
// (conflict-free ds_load_b128 across the 16 rows a fragment touches).
constexpr int W1_STR  = 808;   // 1616 B = 404 dw; 404%64=20
constexpr int W2_STR  = 136;   //  272 B =  68 dw;  68%64=4
constexpr int W3_STR  = 72;    //  144 B =  36 dw;  36%64=36
constexpr int ADD_STR = 40;    //   80 B =  20 dw;  20%64=20
constexpr int HB_STR  = 136;   // activation buffer stride

struct Offs {
  int w[NUM_ADD + 1];
  int b[NUM_ADD + 1];
  constexpr Offs() : w(), b() {
    int ow = 0, ob = 0;
    for (int i = 0; i < NUM_ADD; ++i) {
      w[i] = ow; b[i] = ob;
      ow += pad16i(SIZES[i + 1]) * ADD_STR;  // each add layer: [pad16(out)][ADD_STR] f16
      ob += pad16i(SIZES[i + 1]);
    }
    w[NUM_ADD] = ow; b[NUM_ADD] = ob;
  }
};
constexpr Offs OFF{};

// ---------------- LDS arena ----------------
constexpr int W1_H   = 0;                        // [128][808] f16
constexpr int W2_H   = W1_H + 128 * W1_STR;      // [64][136]
constexpr int W3_H   = W2_H + 64 * W2_STR;       // [32][72]
constexpr int ADD_H  = W3_H + 32 * W3_STR;       // 30 small layers
constexpr int OUTW_H = ADD_H + OFF.w[NUM_ADD];   // [16][40]
constexpr int HALF_CNT = OUTW_H + 16 * ADD_STR;

constexpr int B1_F   = 0;                        // float bias region
constexpr int B2_F   = B1_F + 128;
constexpr int B3_F   = B2_F + 64;
constexpr int ADDB_F = B3_F + 32;
constexpr int BOUT_F = ADDB_F + OFF.b[NUM_ADD];
constexpr int FLOAT_CNT = BOUT_F + 16;

constexpr int FLOAT_BYTE0 = HALF_CNT * 2;
constexpr int HBUF_BYTE0  = FLOAT_BYTE0 + FLOAT_CNT * 4;
constexpr int HBUF_WAVE_B = 16 * HB_STR * 2;     // per-wave [16][136] f16
constexpr int WAVES       = 8;
constexpr int SMEM_BYTES  = HBUF_BYTE0 + WAVES * HBUF_WAVE_B;
static_assert(SMEM_BYTES <= 320 * 1024, "exceeds WGP LDS");
static_assert((FLOAT_BYTE0 % 16) == 0 && (HBUF_BYTE0 % 16) == 0, "align");
// fc1 DS-immediate check: (t&1)*16*W1_STR*2 + 24*64 + 16 must fit 16 bits
static_assert(16 * W1_STR * 2 + 24 * 64 + 16 < 65536, "DS imm overflow");

union AB { v16h v; v8h h[2]; };

DEVINL v8f wmma_f16(v16h a, v16h b, v8f c) {
  return __builtin_amdgcn_wmma_f32_16x16x32_f16(false, a, false, b, (short)0, c,
                                                false, false);
}

DEVINL v8h cvt8(const float* p) {             // 8 consecutive fp32 -> 8 f16
  v4f a = *(const v4f*)p;
  v4f b = *(const v4f*)(p + 4);
  v8h r;
  r[0] = (_Float16)a[0]; r[1] = (_Float16)a[1];
  r[2] = (_Float16)a[2]; r[3] = (_Float16)a[3];
  r[4] = (_Float16)b[0]; r[5] = (_Float16)b[1];
  r[6] = (_Float16)b[2]; r[7] = (_Float16)b[3];
  return r;
}

// B fragment (32x16): base pointer + compile-time element offset -> DS imm.
DEVINL v16h loadB_at(const _Float16* bp, int k0) {
  AB b;
  b.h[0] = *(const v8h*)(bp + k0);
  b.h[1] = *(const v8h*)(bp + k0 + 16);
  return b.v;
}
// A fragment (16x32) via pre-hoisted base pointer.
DEVINL v16h loadA_at(const _Float16* ap, int k0) {
  AB a;
  a.h[0] = *(const v8h*)(ap + k0);
  a.h[1] = *(const v8h*)(ap + k0 + 16);
  return a.v;
}

// Generic fused layer: Hout = act(Hin[16][KC*32] @ W^T + bias)
template<int NT, int KC, bool RELU>
DEVINL void layer_lds(const _Float16* Hin, const _Float16* W, int wstr,
                      const float* bias, _Float16* Hout, int lane) {
  const int nl = lane & 15;
  const int ko = (lane >> 4) << 3;            // B: 0 or 8
  const int kb = (lane >> 4) << 3;            // A: 0 or 8
  const _Float16* ap = Hin + nl * HB_STR + kb;
  const _Float16* bp = W + nl * wstr + ko;    // single base; t*16*wstr is DS imm

  v8f acc[NT];
#pragma unroll
  for (int t = 0; t < NT; ++t) acc[t] = v8f{0, 0, 0, 0, 0, 0, 0, 0};
#pragma unroll
  for (int c = 0; c < KC; ++c) {
    v16h a = loadA_at(ap, c * 32);
    v16h bf[NT];
#pragma unroll
    for (int t = 0; t < NT; ++t) bf[t] = loadB_at(bp + t * 16 * wstr, c * 32);
#pragma unroll
    for (int t = 0; t < NT; ++t) acc[t] = wmma_f16(a, bf[t], acc[t]);
  }
  const int mb = (lane >> 4) << 3;            // C layout: M = mb+i, N = nl
#pragma unroll
  for (int t = 0; t < NT; ++t) {
    float bn = bias[t * 16 + nl];
#pragma unroll
    for (int i = 0; i < 8; ++i) {
      float v = acc[t][i] + bn;
      if (RELU) v = fmaxf(v, 0.f);
      Hout[(mb + i) * HB_STR + t * 16 + nl] = (_Float16)v;
    }
  }
}

DEVINL void stageW(_Float16* dst, const float* src, int rows, int cols,
                   int padRows, int stride, int tid, int nthr) {
  const int total = padRows * stride;
  for (int idx = tid; idx < total; idx += nthr) {
    int r = idx / stride, c = idx % stride;
    float v = (r < rows && c < cols) ? src[r * cols + c] : 0.f;
    dst[idx] = (_Float16)v;
  }
}
DEVINL void stageB(float* dst, const float* src, int n, int padN, int tid, int nthr) {
  for (int i = tid; i < padN; i += nthr) dst[i] = (i < n) ? src[i] : 0.f;
}

struct Params {
  const float* x;
  const float* W1; const float* b1;
  const float* W2; const float* b2;
  const float* W3; const float* b3;
  const float* addW[NUM_ADD]; const float* addB[NUM_ADD];
  const float* Wout; const float* bout;
  float* out;
  int batch;
};

__global__ __launch_bounds__(256, 1)
__attribute__((amdgpu_waves_per_eu(2)))
void mlp_fused(Params p) {
  extern __shared__ char smem[];
  _Float16* Wh = (_Float16*)smem;
  float*    Fl = (float*)(smem + FLOAT_BYTE0);
  const int tid = threadIdx.x, nthr = blockDim.x;

  // ---- stage all weights once (fp32 -> f16, zero-padded) ----
  stageW(Wh + W1_H, p.W1, 128, IN_DIM, 128, W1_STR, tid, nthr);
  stageW(Wh + W2_H, p.W2, 64, 128, 64, W2_STR, tid, nthr);
  stageW(Wh + W3_H, p.W3, 32, 64, 32, W3_STR, tid, nthr);
#pragma unroll 1
  for (int i = 0; i < NUM_ADD; ++i)
    stageW(Wh + ADD_H + OFF.w[i], p.addW[i], SIZES[i + 1], SIZES[i],
           pad16i(SIZES[i + 1]), ADD_STR, tid, nthr);
  stageW(Wh + OUTW_H, p.Wout, 10, 10, 16, ADD_STR, tid, nthr);
  stageB(Fl + B1_F, p.b1, 128, 128, tid, nthr);
  stageB(Fl + B2_F, p.b2, 64, 64, tid, nthr);
  stageB(Fl + B3_F, p.b3, 32, 32, tid, nthr);
#pragma unroll 1
  for (int i = 0; i < NUM_ADD; ++i)
    stageB(Fl + ADDB_F + OFF.b[i], p.addB[i], SIZES[i + 1],
           pad16i(SIZES[i + 1]), tid, nthr);
  stageB(Fl + BOUT_F, p.bout, 10, 16, tid, nthr);
  __syncthreads();

  const int wave = tid >> 5, lane = tid & 31;
  _Float16* hbuf = (_Float16*)(smem + HBUF_BYTE0 + wave * HBUF_WAVE_B);
  const int nTiles = p.batch >> 5;                       // 32 rows per wave-tile
  const int step   = gridDim.x * WAVES;

  const int nl  = lane & 15;
  const int hi8 = (lane >> 4) << 3;                      // 0 or 8
  const int ko  = (lane >> 4) << 3;                      // 0 or 8

  // fc1 B bases: 4 pointers at n-rows {0,32,64,96}; pair member + k-chunk
  // offsets are compile-time -> every fc1 ds_load uses an immediate.
  const _Float16* bp4[4];
#pragma unroll
  for (int j = 0; j < 4; ++j) bp4[j] = Wh + W1_H + (j * 32 + nl) * W1_STR + ko;

  // fc2..out chain for one 16-row half whose fc1 accumulators are in `acc`
  auto tail_net = [&](const v8f (&acc)[8], int row0) {
#pragma unroll
    for (int t = 0; t < 8; ++t) {
      float bn = Fl[B1_F + t * 16 + nl];
#pragma unroll
      for (int i = 0; i < 8; ++i) {
        float v = fmaxf(acc[t][i] + bn, 0.f);
        hbuf[(hi8 + i) * HB_STR + t * 16 + nl] = (_Float16)v;  // h1 -> LDS
      }
    }
    layer_lds<4, 4, true >(hbuf, Wh + W2_H, W2_STR, Fl + B2_F, hbuf, lane);
    layer_lds<2, 2, false>(hbuf, Wh + W3_H, W3_STR, Fl + B3_F, hbuf, lane);
#pragma unroll
    for (int i = 0; i < NUM_ADD; ++i) {
      const _Float16* W = Wh + ADD_H + OFF.w[i];
      const float*   bb = Fl + ADDB_F + OFF.b[i];
      if (SIZES[i + 1] > 16) layer_lds<2, 1, true>(hbuf, W, ADD_STR, bb, hbuf, lane);
      else                   layer_lds<1, 1, true>(hbuf, W, ADD_STR, bb, hbuf, lane);
    }
    v8f o = v8f{0, 0, 0, 0, 0, 0, 0, 0};
    v16h a = loadA_at(hbuf + nl * HB_STR + hi8, 0);
    v16h b = loadB_at(Wh + OUTW_H + nl * ADD_STR + ko, 0);
    o = wmma_f16(a, b, o);
    if (nl < 10) {
      float bn = Fl[BOUT_F + nl];
#pragma unroll
      for (int i = 0; i < 8; ++i)
        p.out[(size_t)(row0 + hi8 + i) * 10 + nl] = o[i] + bn;
    }
  };

  for (int tile = blockIdx.x * WAVES + wave; tile < nTiles; tile += step) {
    const float* xm0 = p.x + (size_t)(tile * 32 + nl) * IN_DIM;
    const float* xm1 = xm0 + (size_t)16 * IN_DIM;
    if (tile + step < nTiles)                 // 32 lanes cover 32 next-tile rows
      __builtin_prefetch(p.x + (size_t)((tile + step) * 32 + lane) * IN_DIM, 0, 1);

    // ---- fc1: [32x784] @ [784x128]; each B fragment feeds two WMMAs,
    // ---- with a one-ahead rotating B buffer to hide LDS latency.
    v8f acc0[8], acc1[8];
#pragma unroll
    for (int t = 0; t < 8; ++t) {
      acc0[t] = v8f{0, 0, 0, 0, 0, 0, 0, 0};
      acc1[t] = v8f{0, 0, 0, 0, 0, 0, 0, 0};
    }
#pragma unroll
    for (int c = 0; c < 25; ++c) {            // fully unrolled: k offsets -> imm
      AB a0, a1;
      const float* p0 = xm0 + c * 32 + hi8;
      const float* p1 = xm1 + c * 32 + hi8;
      a0.h[0] = cvt8(p0);
      a1.h[0] = cvt8(p1);
      if (c < 24) { a0.h[1] = cvt8(p0 + 16); a1.h[1] = cvt8(p1 + 16); }
      else        { a0.h[1] = v8h{0,0,0,0,0,0,0,0}; a1.h[1] = v8h{0,0,0,0,0,0,0,0}; }

      v16h bfA = loadB_at(bp4[0], c * 32);    // fragment 0 in flight
#pragma unroll
      for (int t = 0; t < 8; ++t) {
        v16h bfN = bfA;
        if (t < 7)                            // issue t+1 before consuming t
          bfN = loadB_at(bp4[(t + 1) >> 1] + ((t + 1) & 1) * 16 * W1_STR, c * 32);
        acc0[t] = wmma_f16(a0.v, bfA, acc0[t]); // one LDS fragment ...
        acc1[t] = wmma_f16(a1.v, bfA, acc1[t]); // ... feeds two WMMAs
        bfA = bfN;
      }
    }
    tail_net(acc0, tile * 32);
    tail_net(acc1, tile * 32 + 16);
  }
}

extern "C" void kernel_launch(void* const* d_in, const int* in_sizes, int n_in,
                              void* d_out, int out_size, void* d_ws, size_t ws_size,
                              hipStream_t stream) {
  (void)d_ws; (void)ws_size; (void)out_size;
  Params P;
  P.x  = (const float*)d_in[0];
  P.W1 = (const float*)d_in[1]; P.b1 = (const float*)d_in[2];
  P.W2 = (const float*)d_in[3]; P.b2 = (const float*)d_in[4];
  P.W3 = (const float*)d_in[5]; P.b3 = (const float*)d_in[6];
  if (n_in >= 7 + 2 * NUM_ADD + 2) {          // lists expanded to one ptr each
    for (int i = 0; i < NUM_ADD; ++i) P.addW[i] = (const float*)d_in[7 + i];
    for (int i = 0; i < NUM_ADD; ++i) P.addB[i] = (const float*)d_in[7 + NUM_ADD + i];
    P.Wout = (const float*)d_in[7 + 2 * NUM_ADD];
    P.bout = (const float*)d_in[7 + 2 * NUM_ADD + 1];
  } else {                                    // lists concatenated flat
    const float* aw = (const float*)d_in[7];
    const float* ab = (const float*)d_in[8];
    size_t ow = 0, ob = 0;
    for (int i = 0; i < NUM_ADD; ++i) {
      P.addW[i] = aw + ow; ow += (size_t)SIZES[i + 1] * SIZES[i];
      P.addB[i] = ab + ob; ob += (size_t)SIZES[i + 1];
    }
    P.Wout = (const float*)d_in[9];
    P.bout = (const float*)d_in[10];
  }
  P.out   = (float*)d_out;
  P.batch = in_sizes[0] / IN_DIM;

  // Persistent grid: one 320KB workgroup per WGP. multiProcessorCount reports
  // CUs; a WGP is 2 CUs. Cap so every block has work (batch/32/WAVES tiles).
  int cu = 0;
  (void)hipDeviceGetAttribute(&cu, hipDeviceAttributeMultiprocessorCount, 0);
  int nwg = (cu > 0) ? (cu / 2) : 256;
  if (nwg < 32) nwg = 32;
  int maxwg = (P.batch / 32) / WAVES;         // 512 for batch 131072
  if (nwg > maxwg) nwg = maxwg;

  (void)hipFuncSetAttribute((const void*)mlp_fused,
                            hipFuncAttributeMaxDynamicSharedMemorySize,
                            SMEM_BYTES);
  mlp_fused<<<dim3(nwg), dim3(256), SMEM_BYTES, stream>>>(P);
}
